// CharmGNN_68470368633060
// MI455X (gfx1250) — matrix operand order repaired and checked by reading, not compiled, not run on hardware
//
#include <hip/hip_runtime.h>
#include <hip/hip_bf16.h>
#include <stdint.h>

typedef __attribute__((ext_vector_type(16))) __bf16 v16bf;
typedef __attribute__((ext_vector_type(8)))  __bf16 v8bf;
typedef __attribute__((ext_vector_type(8)))  float  v8f;
typedef __attribute__((ext_vector_type(4)))  int    v4i;

static constexpr int kB = 4, kP = 2048, kQ = 1024, kQN = 32, kD = 256, kH = 128, kSteps = 2;

#if __has_builtin(__builtin_amdgcn_global_load_async_to_lds_b128) && \
    __has_builtin(__builtin_amdgcn_s_wait_asynccnt)
#define USE_ASYNC_LDS 1
#else
#define USE_ASYNC_LDS 0
#endif

// ---------------------------------------------------------------------------
// Adjacency prep: masked bf16 adjacency + row sums. grid=(N,B), block=256
// ---------------------------------------------------------------------------
__global__ __launch_bounds__(256) void prep_adj_kernel(
    const int* __restrict__ graph, const int* __restrict__ mask,
    __bf16* __restrict__ adj, float* __restrict__ rowsum, int N) {
  int b = blockIdx.y, i = blockIdx.x;
  const int* grow = graph + ((size_t)b * N + (size_t)i) * N;
  const int* m = mask + (size_t)b * N;
  __bf16* arow = adj + ((size_t)b * N + (size_t)i) * N;
  float mi = (float)m[i];
  float s = 0.f;
  for (int j = threadIdx.x; j < N; j += 256) {
    float v = mi * (float)m[j] * (float)grow[j];
    arow[j] = (__bf16)v;
    s += v;
  }
  __shared__ float red[256];
  red[threadIdx.x] = s; __syncthreads();
  for (int st = 128; st > 0; st >>= 1) {
    if (threadIdx.x < st) red[threadIdx.x] += red[threadIdx.x + st];
    __syncthreads();
  }
  if (threadIdx.x == 0) rowsum[(size_t)b * N + i] = red[0];
}

// grid=B, block=256: cnt[b] = sum(mask[b,:])
__global__ __launch_bounds__(256) void mask_count_kernel(
    const int* __restrict__ mask, float* __restrict__ cnt, int N) {
  int b = blockIdx.x;
  float s = 0.f;
  for (int j = threadIdx.x; j < N; j += 256) s += (float)mask[(size_t)b * N + j];
  __shared__ float red[256];
  red[threadIdx.x] = s; __syncthreads();
  for (int st = 128; st > 0; st >>= 1) {
    if (threadIdx.x < st) red[threadIdx.x] += red[threadIdx.x + st];
    __syncthreads();
  }
  if (threadIdx.x == 0) cnt[b] = red[0];
}

__global__ void finalize_pnb_kernel(float* nb, long long n) {
  long long i = (long long)blockIdx.x * blockDim.x + threadIdx.x;
  if (i < n) { float v = nb[i]; nb[i] = (v >= 1.f) ? v : 1.f; }
}

__global__ void finalize_qnb_kernel(float* __restrict__ nb, const int* __restrict__ qmask,
                                    const float* __restrict__ cntP, int Q, long long n) {
  long long i = (long long)blockIdx.x * blockDim.x + threadIdx.x;
  if (i < n) {
    int b = (int)(i / Q);
    float v = nb[i] + (qmask[i] ? cntP[b] : 0.f);
    nb[i] = (v >= 1.f) ? v : 1.f;
  }
}

__global__ void compute_qunb_kernel(float* __restrict__ nb, const int* __restrict__ qumask,
                                    const float* __restrict__ cntP, const float* __restrict__ cntQ,
                                    int QN, long long n) {
  long long i = (long long)blockIdx.x * blockDim.x + threadIdx.x;
  if (i < n) {
    int b = (int)(i / QN);
    float v = qumask[i] ? (cntP[b] + cntQ[b]) : 0.f;
    nb[i] = (v >= 1.f) ? v : 1.f;
  }
}

// ---------------------------------------------------------------------------
// Node gate weights: one wave32 per node. w = sigmoid(x . W_node + b)
// ---------------------------------------------------------------------------
__global__ __launch_bounds__(256) void node_weight_kernel(
    const float* __restrict__ X, const float* __restrict__ Wn, const float* __restrict__ bn,
    float* __restrict__ w, long long nNodes, int D) {
  int lane = threadIdx.x & 31;
  long long node = (long long)blockIdx.x * 8 + (threadIdx.x >> 5);
  if (node >= nNodes) return;
  const float* x = X + node * D;
  float s = 0.f;
  for (int d = lane; d < D; d += 32) s += x[d] * Wn[d];
  for (int off = 16; off > 0; off >>= 1) s += __shfl_xor(s, off, 32);
  if (lane == 0) w[node] = 1.f / (1.f + __expf(-(s + bn[0])));
}

// ---------------------------------------------------------------------------
// r[b,d] = sum_i mask[b,i]*w[b,i]*X[b,i,d]  (deterministic partial reduction)
// ---------------------------------------------------------------------------
__global__ __launch_bounds__(256) void mw_sum_partial_kernel(
    const float* __restrict__ X, const int* __restrict__ mask, const float* __restrict__ w,
    float* __restrict__ rpart, int Nn, int D, int chunks) {
  int b = blockIdx.x, c = blockIdx.y, d = threadIdx.x;
  int per = (Nn + chunks - 1) / chunks;
  int i0 = c * per, i1 = i0 + per; if (i1 > Nn) i1 = Nn;
  float s = 0.f;
  for (int i = i0; i < i1; ++i) {
    if (mask[(size_t)b * Nn + i])
      s += w[(size_t)b * Nn + i] * X[((size_t)b * Nn + i) * D + d];
  }
  rpart[((size_t)b * chunks + c) * D + d] = s;
}

__global__ __launch_bounds__(256) void mw_sum_reduce_kernel(
    const float* __restrict__ rpart, float* __restrict__ r, int D, int chunks) {
  int b = blockIdx.x, d = threadIdx.x;
  float s = 0.f;
  for (int c = 0; c < chunks; ++c) s += rpart[((size_t)b * chunks + c) * D + d];
  r[(size_t)b * D + d] = s;
}

// out[b,n] = sum_k r[b,k]*W[k,n].  grid=B, block=D
__global__ __launch_bounds__(256) void small_gemv_kernel(
    const float* __restrict__ r, const float* __restrict__ W, float* __restrict__ out, int D) {
  int b = blockIdx.x, n = threadIdx.x;
  float s = 0.f;
  for (int k = 0; k < D; ++k) s += r[(size_t)b * D + k] * W[(size_t)k * D + n];
  out[(size_t)b * D + n] = s;
}

// ---------------------------------------------------------------------------
// Conversions
// ---------------------------------------------------------------------------
__global__ void f32_to_bf16_kernel(const float* __restrict__ src, __bf16* __restrict__ dst,
                                   long long n) {
  long long i = (long long)blockIdx.x * blockDim.x + threadIdx.x;
  if (i < n) dst[i] = (__bf16)src[i];
}

// src [K,N] f32 -> dst [N,K] bf16
__global__ void transpose_bf16_kernel(const float* __restrict__ src, __bf16* __restrict__ dst,
                                      int K, int N) {
  long long i = (long long)blockIdx.x * blockDim.x + threadIdx.x;
  long long n = (long long)K * N;
  if (i < n) {
    int k = (int)(i / N), c = (int)(i % N);
    dst[(size_t)c * K + k] = (__bf16)src[i];
  }
}

// ---------------------------------------------------------------------------
// WMMA GEMM: C[M,N] = A[M,K] @ B[K,N], A bf16 row-major, B supplied transposed
// (Bt[N][K], bf16).  mode 0: f32 C (+bias).  mode 1: bf16 Ct[N][M] with
// per-row scale (produces the pw-scaled transposed feature matrix that is the
// B operand of the following aggregation GEMM).
//
// 256 threads = 8 wave32, 128x128 block tile, 32x64 per-wave tile
// (2 M-tiles x 4 N-tiles = 8 x v_wmma_f32_16x16x32_bf16 per K-slab).
// Double-buffered LDS staging; async-to-LDS path (ASYNCcnt protocol) when the
// toolchain exposes the gfx1250 builtins, classic load+ds_store otherwise.
// LDS rows padded to 40 elems (80B = 20 banks; gcd(20,64)=4 -> conflict-free
// 16-lane fragment reads).
// ---------------------------------------------------------------------------
#define BM 128
#define BN 128
#define BK 32
#define LDT 40

__global__ __launch_bounds__(256) void wmma_gemm_kernel(
    const __bf16* __restrict__ A, long long sAb, int lda,
    const __bf16* __restrict__ Bt, long long sBb, int ldb,
    float* __restrict__ C, long long sCb, int ldc, const float* __restrict__ bias,
    __bf16* __restrict__ Ct, long long sCtb, int ldct,
    const float* __restrict__ scale, long long sScb,
    int M, int N, int K, int mode) {
  __shared__ __bf16 As[2][BM][LDT];
  __shared__ __bf16 Bs[2][BN][LDT];

  const int tid = threadIdx.x;
  const int lane = tid & 31;
  const int wave = tid >> 5;
  const int wm = wave >> 1;            // 0..3 -> 32-row slice
  const int wn = wave & 1;             // 0..1 -> 64-col slice
  const int bz = blockIdx.z;
  const int mblk = blockIdx.y * BM;
  const int nblk = blockIdx.x * BN;
  const int mlo = lane & 15, half = lane >> 4;

  const __bf16* Ab = A + (long long)bz * sAb;
  const __bf16* Bb = Bt + (long long)bz * sBb;

  // staging: each thread copies 16 bf16 (2x16B) of A and of B per K-slab
  const int srow = tid >> 1;           // 0..127
  const int skc  = (tid & 1) * 16;     // 0 or 16
  int arow = mblk + srow; if (arow >= M) arow = M - 1;  // clamp, stores guarded
  const __bf16* ag = Ab + (long long)arow * lda + skc;
  const __bf16* bg = Bb + (long long)(nblk + srow) * ldb + skc;

  auto stage = [&](int kk, int buf) {
#if USE_ASYNC_LDS
    typedef __attribute__((address_space(1))) v4i gv4i;  // printed as "v4i __device__*"
    typedef __attribute__((address_space(3))) v4i lv4i;
    __builtin_amdgcn_global_load_async_to_lds_b128(
        (gv4i*)(uintptr_t)(ag + kk),
        (lv4i*)(unsigned)(uintptr_t)&As[buf][srow][skc], 0, 0);
    __builtin_amdgcn_global_load_async_to_lds_b128(
        (gv4i*)(uintptr_t)(ag + kk + 8),
        (lv4i*)(unsigned)(uintptr_t)&As[buf][srow][skc + 8], 0, 0);
    __builtin_amdgcn_global_load_async_to_lds_b128(
        (gv4i*)(uintptr_t)(bg + kk),
        (lv4i*)(unsigned)(uintptr_t)&Bs[buf][srow][skc], 0, 0);
    __builtin_amdgcn_global_load_async_to_lds_b128(
        (gv4i*)(uintptr_t)(bg + kk + 8),
        (lv4i*)(unsigned)(uintptr_t)&Bs[buf][srow][skc + 8], 0, 0);
#else
    v8bf a0 = *(const v8bf*)(ag + kk);
    v8bf a1 = *(const v8bf*)(ag + kk + 8);
    v8bf b0 = *(const v8bf*)(bg + kk);
    v8bf b1 = *(const v8bf*)(bg + kk + 8);
    *(v8bf*)&As[buf][srow][skc]     = a0;
    *(v8bf*)&As[buf][srow][skc + 8] = a1;
    *(v8bf*)&Bs[buf][srow][skc]     = b0;
    *(v8bf*)&Bs[buf][srow][skc + 8] = b1;
#endif
    // speculative prefetch of the slab after next (global_prefetch_b8)
    __builtin_prefetch(ag + kk + BK, 0, 1);
    __builtin_prefetch(bg + kk + BK, 0, 1);
  };

  v8f acc[2][4] = {};
  const int nk = K / BK;

  stage(0, 0);
  for (int ks = 0; ks < nk; ++ks) {
    const int cur = ks & 1;
#if USE_ASYNC_LDS
    __builtin_amdgcn_s_wait_asynccnt(0);   // own stage(ks) complete
#endif
    __syncthreads();                       // all waves' stage(ks) complete;
                                           // all reads of buf cur^1 retired
    if (ks + 1 < nk) stage((ks + 1) * BK, cur ^ 1);

    v16bf af[2];
#pragma unroll
    for (int i = 0; i < 2; ++i) {
      int lm = wm * 32 + i * 16 + mlo;
      union { v16bf v; v8bf h[2]; } u;
      u.h[0] = *(const v8bf*)&As[cur][lm][half * 8];        // K = half*8..+7
      u.h[1] = *(const v8bf*)&As[cur][lm][16 + half * 8];   // K = 16+half*8..+7
      af[i] = u.v;
    }
#pragma unroll
    for (int j = 0; j < 4; ++j) {
      int ln = wn * 64 + j * 16 + mlo;
      union { v16bf v; v8bf h[2]; } u;
      u.h[0] = *(const v8bf*)&Bs[cur][ln][half * 8];        // K = half*8..+7
      u.h[1] = *(const v8bf*)&Bs[cur][ln][16 + half * 8];   // K = 16+half*8..+7
      v16bf bfr = u.v;
#pragma unroll
      for (int i = 0; i < 2; ++i)
        acc[i][j] = __builtin_amdgcn_wmma_f32_16x16x32_bf16(
            false, af[i], false, bfr, (short)0, acc[i][j], false, false);
    }
  }

  if (mode == 0) {
    float* Cb = C + (long long)bz * sCb;
#pragma unroll
    for (int i = 0; i < 2; ++i) {
      int rbase = mblk + wm * 32 + i * 16 + half * 8;       // VGPR r <-> row rbase+r
#pragma unroll
      for (int j = 0; j < 4; ++j) {
        int col = nblk + wn * 64 + j * 16 + mlo;
        float bv = bias ? bias[col] : 0.f;
#pragma unroll
        for (int r = 0; r < 8; ++r) {
          int row = rbase + r;
          if (row < M) Cb[(long long)row * ldc + col] = acc[i][j][r] + bv;
        }
      }
    }
  } else {
    __bf16* Cb = Ct + (long long)bz * sCtb;
    const float* sc = scale + (long long)bz * sScb;
#pragma unroll
    for (int i = 0; i < 2; ++i) {
      int rbase = mblk + wm * 32 + i * 16 + half * 8;
#pragma unroll
      for (int j = 0; j < 4; ++j) {
        int col = nblk + wn * 64 + j * 16 + mlo;
        v8bf outv;
#pragma unroll
        for (int r = 0; r < 8; ++r) outv[r] = (__bf16)(acc[i][j][r] * sc[rbase + r]);
        *(v8bf*)(Cb + (long long)col * ldct + rbase) = outv;  // contiguous 16B
      }
    }
  }
}

// ---------------------------------------------------------------------------
// Update kernels
// ---------------------------------------------------------------------------
__global__ void update_p_kernel(const float* __restrict__ selfp, const float* __restrict__ info,
                                const float* __restrict__ nb, float* __restrict__ out,
                                int D, long long total) {
  long long i = (long long)blockIdx.x * blockDim.x + threadIdx.x;
  if (i < total) {
    long long node = i / D;
    float v = selfp[i] + info[i] / nb[node];
    out[i] = v > 0.f ? v : 0.f;
  }
}

__global__ void update_q_kernel(const float* __restrict__ selfq, const float* __restrict__ info,
                                const float* __restrict__ s_qd, const int* __restrict__ qmask,
                                const float* __restrict__ nb, float* __restrict__ out,
                                int Q, int D, long long total) {
  long long i = (long long)blockIdx.x * blockDim.x + threadIdx.x;
  if (i < total) {
    long long node = i / D; int d = (int)(i % D); int b = (int)(node / Q);
    float extra = qmask[node] ? s_qd[(size_t)b * D + d] : 0.f;
    float v = selfq[i] + (extra + info[i]) / nb[node];
    out[i] = v > 0.f ? v : 0.f;
  }
}

__global__ void update_qu_kernel(const float* __restrict__ selfqu, const float* __restrict__ s_qup,
                                 const float* __restrict__ s_quq, const int* __restrict__ qumask,
                                 const float* __restrict__ nb, float* __restrict__ out,
                                 int QN, int D, long long total) {
  long long i = (long long)blockIdx.x * blockDim.x + threadIdx.x;
  if (i < total) {
    long long node = i / D; int d = (int)(i % D); int b = (int)(node / QN);
    float extra = qumask[node] ? (s_qup[(size_t)b * D + d] + s_quq[(size_t)b * D + d]) : 0.f;
    float v = selfqu[i] + extra / nb[node];
    out[i] = v > 0.f ? v : 0.f;
  }
}

// ---------------------------------------------------------------------------
// Host orchestration
// ---------------------------------------------------------------------------
static inline void launch_gemm(hipStream_t s,
                               const __bf16* A, long long sAb, int lda,
                               const __bf16* Bt, long long sBb, int ldb,
                               float* C, long long sCb, int ldc, const float* bias,
                               __bf16* Ct, long long sCtb, int ldct,
                               const float* scale, long long sScb,
                               int M, int N, int K, int batch, int mode) {
  dim3 grid(N / BN, (M + BM - 1) / BM, batch), block(256);
  wmma_gemm_kernel<<<grid, block, 0, s>>>(A, sAb, lda, Bt, sBb, ldb, C, sCb, ldc, bias,
                                          Ct, sCtb, ldct, scale, sScb, M, N, K, mode);
}

extern "C" void kernel_launch(void* const* d_in, const int* in_sizes, int n_in,
                              void* d_out, int out_size, void* d_ws, size_t ws_size,
                              hipStream_t stream) {
  (void)in_sizes; (void)n_in; (void)out_size; (void)ws_size;
  const float* p_node   = (const float*)d_in[0];
  const float* q_node   = (const float*)d_in[1];
  const float* qu_node  = (const float*)d_in[2];
  const int*   p_mask   = (const int*)d_in[3];
  const int*   q_mask   = (const int*)d_in[4];
  const int*   qu_mask  = (const int*)d_in[5];
  const int*   pp_graph = (const int*)d_in[6];
  const int*   qq_graph = (const int*)d_in[7];
  const float* W_node   = (const float*)d_in[10];
  const float* b_node   = (const float*)d_in[11];
  const float* W_self   = (const float*)d_in[12];
  const float* b_self   = (const float*)d_in[13];
  const float* W_pp     = (const float*)d_in[14];
  const float* W_qd     = (const float*)d_in[15];
  const float* W_qq     = (const float*)d_in[16];
  const float* W_qup    = (const float*)d_in[17];
  const float* W_quq    = (const float*)d_in[18];
  const float* W_p      = (const float*)d_in[19];
  const float* b_p      = (const float*)d_in[20];
  const float* W_q      = (const float*)d_in[21];
  const float* b_q      = (const float*)d_in[22];
  const float* W_qu     = (const float*)d_in[23];
  const float* b_qu     = (const float*)d_in[24];

  char* ws = (char*)d_ws;
  size_t off = 0;
  auto alloc = [&](size_t bytes) -> char* {
    char* p = ws + off;
    off += (bytes + 255) & ~(size_t)255;
    return p;
  };
  const int CH = 16;  // mw_sum chunks
  __bf16* PPb   = (__bf16*)alloc((size_t)kB * kP * kP * 2);
  __bf16* QQb   = (__bf16*)alloc((size_t)kB * kQ * kQ * 2);
  float* p_nb   = (float*)alloc((size_t)kB * kP * 4);
  float* q_nb   = (float*)alloc((size_t)kB * kQ * 4);
  float* qu_nb  = (float*)alloc((size_t)kB * kQN * 4);
  float* cntP   = (float*)alloc((size_t)kB * 4);
  float* cntQ   = (float*)alloc((size_t)kB * 4);
  float* pw     = (float*)alloc((size_t)kB * kP * 4);
  float* qw     = (float*)alloc((size_t)kB * kQ * 4);
  float* rpart  = (float*)alloc((size_t)kB * CH * kD * 4);
  float* r_p    = (float*)alloc((size_t)kB * kD * 4);
  float* r_q    = (float*)alloc((size_t)kB * kD * 4);
  float* s_qd   = (float*)alloc((size_t)kB * kD * 4);
  float* s_qup  = (float*)alloc((size_t)kB * kD * 4);
  float* s_quq  = (float*)alloc((size_t)kB * kD * 4);
  __bf16* Wt_self = (__bf16*)alloc((size_t)kD * kD * 2);
  __bf16* Wt_pp   = (__bf16*)alloc((size_t)kD * kD * 2);
  __bf16* Wt_qq   = (__bf16*)alloc((size_t)kD * kD * 2);
  __bf16* Wt_p    = (__bf16*)alloc((size_t)kD * kH * 2);
  __bf16* Wt_q    = (__bf16*)alloc((size_t)kD * kH * 2);
  __bf16* Wt_qu   = (__bf16*)alloc((size_t)kD * kH * 2);
  __bf16* pb    = (__bf16*)alloc((size_t)kB * kP * kD * 2);
  __bf16* qb    = (__bf16*)alloc((size_t)kB * kQ * kD * 2);
  __bf16* qub   = (__bf16*)alloc((size_t)kB * kQN * kD * 2);
  float* self_p = (float*)alloc((size_t)kB * kP * kD * 4);
  float* self_q = (float*)alloc((size_t)kB * kQ * kD * 4);
  float* self_qu= (float*)alloc((size_t)kB * kQN * kD * 4);
  __bf16* Ypp   = (__bf16*)alloc((size_t)kB * kD * kP * 2);   // [B][D][P]
  __bf16* Yqq   = (__bf16*)alloc((size_t)kB * kD * kQ * 2);   // [B][D][Q]
  float* pp_info= (float*)alloc((size_t)kB * kP * kD * 4);
  float* qq_info= (float*)alloc((size_t)kB * kQ * kD * 4);
  float* p_buf  = (float*)alloc((size_t)kB * kP * kD * 4);
  float* q_buf  = (float*)alloc((size_t)kB * kQ * kD * 4);
  float* qu_buf = (float*)alloc((size_t)kB * kQN * kD * 4);

  // ---- step-invariant prep -------------------------------------------------
  prep_adj_kernel<<<dim3(kP, kB), 256, 0, stream>>>(pp_graph, p_mask, PPb, p_nb, kP);
  prep_adj_kernel<<<dim3(kQ, kB), 256, 0, stream>>>(qq_graph, q_mask, QQb, q_nb, kQ);
  mask_count_kernel<<<kB, 256, 0, stream>>>(p_mask, cntP, kP);
  mask_count_kernel<<<kB, 256, 0, stream>>>(q_mask, cntQ, kQ);
  finalize_pnb_kernel<<<(kB * kP + 255) / 256, 256, 0, stream>>>(p_nb, (long long)kB * kP);
  finalize_qnb_kernel<<<(kB * kQ + 255) / 256, 256, 0, stream>>>(q_nb, q_mask, cntP, kQ,
                                                                 (long long)kB * kQ);
  compute_qunb_kernel<<<(kB * kQN + 255) / 256, 256, 0, stream>>>(qu_nb, qu_mask, cntP, cntQ,
                                                                  kQN, (long long)kB * kQN);
  transpose_bf16_kernel<<<(kD * kD + 255) / 256, 256, 0, stream>>>(W_self, Wt_self, kD, kD);
  transpose_bf16_kernel<<<(kD * kD + 255) / 256, 256, 0, stream>>>(W_pp, Wt_pp, kD, kD);
  transpose_bf16_kernel<<<(kD * kD + 255) / 256, 256, 0, stream>>>(W_qq, Wt_qq, kD, kD);
  transpose_bf16_kernel<<<(kD * kH + 255) / 256, 256, 0, stream>>>(W_p, Wt_p, kD, kH);
  transpose_bf16_kernel<<<(kD * kH + 255) / 256, 256, 0, stream>>>(W_q, Wt_q, kD, kH);
  transpose_bf16_kernel<<<(kD * kH + 255) / 256, 256, 0, stream>>>(W_qu, Wt_qu, kD, kH);

  const float* pcur = p_node;
  const float* qcur = q_node;
  const float* qucur = qu_node;

  for (int step = 0; step < kSteps; ++step) {
    // gate weights
    node_weight_kernel<<<(kB * kP + 7) / 8, 256, 0, stream>>>(pcur, W_node, b_node, pw,
                                                              (long long)kB * kP, kD);
    node_weight_kernel<<<(kB * kQ + 7) / 8, 256, 0, stream>>>(qcur, W_node, b_node, qw,
                                                              (long long)kB * kQ, kD);
    // rank-1 reductions: r = sum(mask*w*x)
    mw_sum_partial_kernel<<<dim3(kB, CH), kD, 0, stream>>>(pcur, p_mask, pw, rpart, kP, kD, CH);
    mw_sum_reduce_kernel<<<kB, kD, 0, stream>>>(rpart, r_p, kD, CH);
    mw_sum_partial_kernel<<<dim3(kB, CH), kD, 0, stream>>>(qcur, q_mask, qw, rpart, kQ, kD, CH);
    mw_sum_reduce_kernel<<<kB, kD, 0, stream>>>(rpart, r_q, kD, CH);
    small_gemv_kernel<<<kB, kD, 0, stream>>>(r_p, W_qd, s_qd, kD);
    small_gemv_kernel<<<kB, kD, 0, stream>>>(r_p, W_qup, s_qup, kD);
    small_gemv_kernel<<<kB, kD, 0, stream>>>(r_q, W_quq, s_quq, kD);

    // bf16 node copies
    f32_to_bf16_kernel<<<((long long)kB * kP * kD + 255) / 256, 256, 0, stream>>>(
        pcur, pb, (long long)kB * kP * kD);
    f32_to_bf16_kernel<<<((long long)kB * kQ * kD + 255) / 256, 256, 0, stream>>>(
        qcur, qb, (long long)kB * kQ * kD);
    f32_to_bf16_kernel<<<((long long)kB * kQN * kD + 255) / 256, 256, 0, stream>>>(
        qucur, qub, (long long)kB * kQN * kD);

    // self transforms (f32 out, +bias)
    launch_gemm(stream, pb, (long long)kP * kD, kD, Wt_self, 0, kD,
                self_p, (long long)kP * kD, kD, b_self, nullptr, 0, 0, nullptr, 0,
                kP, kD, kD, kB, 0);
    launch_gemm(stream, qb, (long long)kQ * kD, kD, Wt_self, 0, kD,
                self_q, (long long)kQ * kD, kD, b_self, nullptr, 0, 0, nullptr, 0,
                kQ, kD, kD, kB, 0);
    launch_gemm(stream, qub, (long long)kQN * kD, kD, Wt_self, 0, kD,
                self_qu, (long long)kQN * kD, kD, b_self, nullptr, 0, 0, nullptr, 0,
                kQN, kD, kD, kB, 0);

    // pw-scaled transposed features (mode 1 -> bf16 [D][P] / [D][Q])
    launch_gemm(stream, pb, (long long)kP * kD, kD, Wt_pp, 0, kD,
                nullptr, 0, 0, nullptr, Ypp, (long long)kD * kP, kP, pw, kP,
                kP, kD, kD, kB, 1);
    launch_gemm(stream, qb, (long long)kQ * kD, kD, Wt_qq, 0, kD,
                nullptr, 0, 0, nullptr, Yqq, (long long)kD * kQ, kQ, qw, kQ,
                kQ, kD, kD, kB, 1);

    // adjacency aggregation GEMMs (the heavy ones: K=P / K=Q)
    launch_gemm(stream, PPb, (long long)kP * kP, kP, Ypp, (long long)kD * kP, kP,
                pp_info, (long long)kP * kD, kD, nullptr, nullptr, 0, 0, nullptr, 0,
                kP, kD, kP, kB, 0);
    launch_gemm(stream, QQb, (long long)kQ * kQ, kQ, Yqq, (long long)kD * kQ, kQ,
                qq_info, (long long)kQ * kD, kD, nullptr, nullptr, 0, 0, nullptr, 0,
                kQ, kD, kQ, kB, 0);

    // updates
    update_p_kernel<<<((long long)kB * kP * kD + 255) / 256, 256, 0, stream>>>(
        self_p, pp_info, p_nb, p_buf, kD, (long long)kB * kP * kD);
    update_q_kernel<<<((long long)kB * kQ * kD + 255) / 256, 256, 0, stream>>>(
        self_q, qq_info, s_qd, q_mask, q_nb, q_buf, kQ, kD, (long long)kB * kQ * kD);
    update_qu_kernel<<<((long long)kB * kQN * kD + 255) / 256, 256, 0, stream>>>(
        self_qu, s_qup, s_quq, qu_mask, qu_nb, qu_buf, kQN, kD, (long long)kB * kQN * kD);

    pcur = p_buf; qcur = q_buf; qucur = qu_buf;
  }

  // ---- final projections ---------------------------------------------------
  f32_to_bf16_kernel<<<((long long)kB * kP * kD + 255) / 256, 256, 0, stream>>>(
      pcur, pb, (long long)kB * kP * kD);
  f32_to_bf16_kernel<<<((long long)kB * kQ * kD + 255) / 256, 256, 0, stream>>>(
      qcur, qb, (long long)kB * kQ * kD);
  f32_to_bf16_kernel<<<((long long)kB * kQN * kD + 255) / 256, 256, 0, stream>>>(
      qucur, qub, (long long)kB * kQN * kD);

  float* out = (float*)d_out;
  float* out_p = out;
  float* out_q = out + (size_t)kB * kP * kH;
  float* out_qu = out + (size_t)kB * kP * kH + (size_t)kB * kQ * kH;
  launch_gemm(stream, pb, (long long)kP * kD, kD, Wt_p, 0, kD,
              out_p, (long long)kP * kH, kH, b_p, nullptr, 0, 0, nullptr, 0,
              kP, kH, kD, kB, 0);
  launch_gemm(stream, qb, (long long)kQ * kD, kD, Wt_q, 0, kD,
              out_q, (long long)kQ * kH, kH, b_q, nullptr, 0, 0, nullptr, 0,
              kQ, kH, kD, kB, 0);
  launch_gemm(stream, qub, (long long)kQN * kD, kD, Wt_qu, 0, kD,
              out_qu, (long long)kQN * kH, kH, b_qu, nullptr, 0, 0, nullptr, 0,
              kQN, kH, kD, kB, 0);
}
